// ResGEChebNet_48790828482911
// MI455X (gfx1250) — matrix operand order repaired and checked by reading, not compiled, not run on hardware
//
#include <hip/hip_runtime.h>
#include <math.h>

// ---------------- problem constants ----------------
constexpr int V_NODES = 100000;
constexpr int E_EDGES = 800000;
constexpr int BATCH   = 16;
constexpr int CH      = 16;
constexpr int NCOL    = BATCH * CH;          // 256 feature columns, col = b*16 + c
constexpr int M_ROWS  = V_NODES * BATCH;     // 1,600,000 GEMM rows (m = v*16 + b)
constexpr int TILES   = M_ROWS / 16;         // 100,000 WMMA tiles
constexpr float BN_EPS = 1e-5f;

#define DEVFN static __device__ __forceinline__

// ---------------- bf16 helpers (manual, toolchain-independent) ----------------
DEVFN unsigned short f2bf(float f) {
  unsigned u = __float_as_uint(f);
  u += 0x7fffu + ((u >> 16) & 1u);           // round-to-nearest-even
  return (unsigned short)(u >> 16);
}
DEVFN float bf2f(unsigned short s) { return __uint_as_float(((unsigned)s) << 16); }

DEVFN float4 ld4(const unsigned short* p) {
  ushort4 q = *(const ushort4*)p;            // global_load_b64
  return make_float4(bf2f(q.x), bf2f(q.y), bf2f(q.z), bf2f(q.w));
}
DEVFN void st4(unsigned short* p, float a, float b, float c, float d) {
  ushort4 q; q.x = f2bf(a); q.y = f2bf(b); q.z = f2bf(c); q.w = f2bf(d);
  *(ushort4*)p = q;                          // global_store_b64
}

// ---------------- WMMA vector types ----------------
typedef __attribute__((ext_vector_type(16))) __bf16 v16bf;
typedef __attribute__((ext_vector_type(8)))  float  v8f;

// ============== 1) transpose x (B,C,V) f32 -> xf (V,256) bf16 ==============
__global__ void k_transpose(const float* __restrict__ x, unsigned short* __restrict__ xf) {
  __shared__ float sh[16][17];
  int tx = threadIdx.x & 15, ty = threadIdx.x >> 4;
  int v0 = blockIdx.x * 16, c0 = blockIdx.y * 16;
  sh[ty][tx] = x[(size_t)(c0 + ty) * V_NODES + v0 + tx];   // coalesced over v
  __syncthreads();
  xf[(size_t)(v0 + ty) * NCOL + c0 + tx] = f2bf(sh[tx][ty]); // coalesced over col
}

// ============== 1b) weights f32 -> bf16, once per launch ==============
__global__ void k_wcvt(const float* __restrict__ w, unsigned short* __restrict__ wbf, int n) {
  int i = blockIdx.x * 256 + threadIdx.x;
  if (i < n) wbf[i] = f2bf(w[i]);
}

// ============== 2) CSR-by-destination build ==============
__global__ void k_count(const int* __restrict__ dst, int* __restrict__ cnt) {
  int e = blockIdx.x * 256 + threadIdx.x;
  if (e < E_EDGES) atomicAdd(&cnt[dst[e]], 1);
}

__global__ void k_scan(const int* __restrict__ cnt, int* __restrict__ row_ptr) {
  __shared__ int sh[1024];
  __shared__ int carry;
  int t = threadIdx.x;
  if (t == 0) carry = 0;
  __syncthreads();
  for (int base = 0; base < V_NODES; base += 1024) {
    int i = base + t;
    int val = (i < V_NODES) ? cnt[i] : 0;
    sh[t] = val;
    __syncthreads();
    for (int off = 1; off < 1024; off <<= 1) {
      int add = (t >= off) ? sh[t - off] : 0;
      __syncthreads();
      sh[t] += add;
      __syncthreads();
    }
    int incl = sh[t];
    int cb = carry;
    if (i < V_NODES) row_ptr[i] = cb + incl - val;          // exclusive prefix
    __syncthreads();
    if (t == 1023) carry = cb + incl;
    __syncthreads();
  }
  if (t == 0) row_ptr[V_NODES] = carry;                      // == E_EDGES
}

__global__ void k_fill(const int* __restrict__ src, const int* __restrict__ dst,
                       const float* __restrict__ ew, const int* __restrict__ row_ptr,
                       int* __restrict__ cursor, int* __restrict__ csr_src,
                       float* __restrict__ csr_w) {
  int e = blockIdx.x * 256 + threadIdx.x;
  if (e >= E_EDGES) return;
  int d = dst[e];
  int pos = row_ptr[d] + atomicAdd(&cursor[d], 1);
  csr_src[pos] = src[e];
  csr_w[pos]   = ew[e];
}

// ============== 3) Laplacian apply: y = alpha*(A x - x) + beta*z ==============
// gather-only (no float atomics); 64 lanes x ushort4 cover 256 cols per node.
// bf16 planes keep the 3 live planes (~150 MB) inside the 192 MB L2.
__global__ void k_lap(unsigned short* __restrict__ y, const unsigned short* __restrict__ x,
                      const unsigned short* __restrict__ z, float alpha, float beta,
                      const int* __restrict__ row_ptr, const int* __restrict__ csr_src,
                      const float* __restrict__ csr_w) {
  int node = blockIdx.x * 4 + (threadIdx.x >> 6);
  int col  = (threadIdx.x & 63) * 4;
  if (node >= V_NODES) return;
  float4 xv = ld4(x + (size_t)node * NCOL + col);
  float4 acc = make_float4(-xv.x, -xv.y, -xv.z, -xv.w);
  int e0 = row_ptr[node], e1 = row_ptr[node + 1];
  for (int e = e0; e < e1; ++e) {
    int s = csr_src[e];
    float w = csr_w[e];
    float4 xs = ld4(x + (size_t)s * NCOL + col);   // hot gather, L2-resident
    acc.x = fmaf(w, xs.x, acc.x);
    acc.y = fmaf(w, xs.y, acc.y);
    acc.z = fmaf(w, xs.z, acc.z);
    acc.w = fmaf(w, xs.w, acc.w);
  }
  acc.x *= alpha; acc.y *= alpha; acc.z *= alpha; acc.w *= alpha;
  if (z) {
    float4 zv = ld4(z + (size_t)node * NCOL + col);
    acc.x = fmaf(beta, zv.x, acc.x);
    acc.y = fmaf(beta, zv.y, acc.y);
    acc.z = fmaf(beta, zv.z, acc.z);
    acc.w = fmaf(beta, zv.w, acc.w);
  }
  st4(y + (size_t)node * NCOL + col, acc.x, acc.y, acc.z, acc.w);
}

// ============== 4) Chebyshev combine via WMMA bf16 ==============
// Per wave: one 16-row tile of D[M,16] = [x0|x1|x2|x3][M,64] * W[64,16] (+bias,+res,ReLU)
// Two v_wmma_f32_16x16x32_bf16 per tile (K pairs {x0,x1} and {x2,x3}).
// HAS_RES / DO_RELU are compile-time so the epilogue is branch-free.
template <bool HAS_RES, bool DO_RELU>
__global__ void k_combine(unsigned short* __restrict__ out,
                          const unsigned short* __restrict__ x0,
                          const unsigned short* __restrict__ x1,
                          const unsigned short* __restrict__ x2,
                          const unsigned short* __restrict__ x3,
                          const unsigned short* __restrict__ wbf,  // (4,16,16) bf16 = B[64,16]
                          const float* __restrict__ bias,          // (16,)
                          const unsigned short* __restrict__ res) {
  int wave = (blockIdx.x * blockDim.x + threadIdx.x) >> 5;
  if (wave >= TILES) return;
  int lane = threadIdx.x & 31;
  int hi   = lane >> 4;                 // lane half selects K-subrange / M-subrange
  int mrow = wave * 16 + (lane & 15);   // A-matrix row for this lane

  // A operand, ISA 16-bit A 16x32 layout:
  // lanes 0-15 : elems 0-7 = K0-7   (termA cols 0-7),  elems 8-15 = K16-23 (termB cols 0-7)
  // lanes16-31 : elems 0-7 = K8-15  (termA cols 8-15), elems 8-15 = K24-31 (termB cols 8-15)
  size_t abase = (size_t)mrow * 16 + hi * 8;   // 8 contiguous bf16 = 16B aligned
  union UV { uint4 q[2]; v16bf v; };
  UV a0, a1;
  a0.q[0] = *(const uint4*)(x0 + abase);
  a0.q[1] = *(const uint4*)(x1 + abase);
  a1.q[0] = *(const uint4*)(x2 + abase);
  a1.q[1] = *(const uint4*)(x3 + abase);

  // B operand, 16-bit B 32x16 layout: lane&15 = N column, elements in the same K order as A:
  // elem s<8 -> Wbf[2p][hi*8+s][n], elem s>=8 -> Wbf[2p+1][hi*8+s-8][n]
  UV b0, b1;
#pragma unroll
  for (int s = 0; s < 16; ++s) {
    size_t bo = (size_t)(s >> 3) * 256 + (size_t)(hi * 8 + (s & 7)) * 16 + (lane & 15);
    b0.v[s] = __builtin_bit_cast(__bf16, wbf[bo]);
    b1.v[s] = __builtin_bit_cast(__bf16, wbf[512 + bo]);
  }

  v8f c = {};
  c = __builtin_amdgcn_wmma_f32_16x16x32_bf16(false, a0.v, false, b0.v, (short)0, c, false, false);
  c = __builtin_amdgcn_wmma_f32_16x16x32_bf16(false, a1.v, false, b1.v, (short)0, c, false, false);

  // C/D layout: elem r -> row = wave*16 + hi*8 + r, col = lane&15
  int ocol = lane & 15;
  float bv = bias[ocol];
  size_t ebase = ((size_t)wave * 16 + hi * 8) * 16 + ocol;   // element (r=0), stride 16/r
#pragma unroll
  for (int r = 0; r < 8; ++r) {
    float val = c[r] + bv;
    if (HAS_RES) val += bf2f(res[ebase + (size_t)r * 16]);
    if (DO_RELU) val = fmaxf(val, 0.0f);
    out[ebase + (size_t)r * 16] = f2bf(val);
  }
}

// ============== 5) BatchNorm over (B,V) per channel ==============
__global__ void k_bn_stats(const unsigned short* __restrict__ x, float* __restrict__ stats) {
  int col = threadIdx.x;                       // 256 cols, channel = col & 15
  float s = 0.f, ss = 0.f;
  for (int v = blockIdx.x; v < V_NODES; v += gridDim.x) {
    float val = bf2f(x[(size_t)v * NCOL + col]);
    s += val; ss += val * val;
  }
  __shared__ float sh[256], sh2[256];
  sh[col] = s; sh2[col] = ss;
  __syncthreads();
  if (col < 16) {
    float ts = 0.f, tss = 0.f;
    for (int j = col; j < 256; j += 16) { ts += sh[j]; tss += sh2[j]; }
    atomicAdd(&stats[col], ts);
    atomicAdd(&stats[16 + col], tss);
  }
}

__global__ void k_bn_fin(float* __restrict__ stats, const float* __restrict__ g,
                         const float* __restrict__ b) {
  int c = threadIdx.x;
  if (c >= 16) return;
  float n = (float)V_NODES * (float)BATCH;
  float mean = stats[c] / n;
  float var  = stats[16 + c] / n - mean * mean;
  float sc   = g[c] * rsqrtf(var + BN_EPS);
  stats[32 + c] = sc;
  stats[48 + c] = b[c] - mean * sc;
}

__global__ void k_bn_apply(unsigned short* __restrict__ x, const float* __restrict__ stats) {
  size_t idx = ((size_t)blockIdx.x * 256 + threadIdx.x) * 4;
  if (idx >= (size_t)V_NODES * NCOL) return;
  int c0 = (int)(idx & 15);                    // in {0,4,8,12}; 4 consecutive channels
  float4 v = ld4(x + idx);
  v.x = fmaf(v.x, stats[32 + c0 + 0], stats[48 + c0 + 0]);
  v.y = fmaf(v.y, stats[32 + c0 + 1], stats[48 + c0 + 1]);
  v.z = fmaf(v.z, stats[32 + c0 + 2], stats[48 + c0 + 2]);
  v.w = fmaf(v.w, stats[32 + c0 + 3], stats[48 + c0 + 3]);
  st4(x + idx, v.x, v.y, v.z, v.w);
}

// ============== 6) max-pool over V + softmax over channels ==============
__global__ void k_maxpool(const unsigned short* __restrict__ x, float* __restrict__ pooled) {
  int col = threadIdx.x;
  float m = 0.0f;                              // post-ReLU values are >= 0
  for (int v = blockIdx.x; v < V_NODES; v += gridDim.x)
    m = fmaxf(m, bf2f(x[(size_t)v * NCOL + col]));
  atomicMax((int*)(pooled + col), __float_as_int(m));   // valid for non-negative f32
}

__global__ void k_softmax(const float* __restrict__ pooled, float* __restrict__ out) {
  int b = threadIdx.x;
  if (b >= 16) return;
  float mx = -1e30f;
  for (int c = 0; c < 16; ++c) mx = fmaxf(mx, pooled[b * 16 + c]);
  float e[16], s = 0.f;
  for (int c = 0; c < 16; ++c) { e[c] = expf(pooled[b * 16 + c] - mx); s += e[c]; }
  float inv = 1.0f / s;
  for (int c = 0; c < 16; ++c) out[b * 16 + c] = e[c] * inv;
}

// ---------------- host orchestration ----------------
static inline size_t align256(size_t x) { return (x + 255) & ~(size_t)255; }

extern "C" void kernel_launch(void* const* d_in, const int* in_sizes, int n_in,
                              void* d_out, int out_size, void* d_ws, size_t ws_size,
                              hipStream_t stream) {
  (void)in_sizes; (void)n_in; (void)out_size; (void)ws_size;
  const float* x        = (const float*)d_in[0];   // (B,C,V)
  const float* edge_w   = (const float*)d_in[1];   // (E,)
  const float* weights  = (const float*)d_in[2];   // (6,K,C,C)
  const float* biases   = (const float*)d_in[3];   // (6,C)
  const float* gamma    = (const float*)d_in[4];   // (5,C)
  const float* beta     = (const float*)d_in[5];   // (5,C)
  const int*   edge_src = (const int*)d_in[6];     // (E,)
  const int*   edge_dst = (const int*)d_in[7];     // (E,)
  float*       out      = (float*)d_out;           // (B,C) = 256 floats
  char*        ws       = (char*)d_ws;

  const size_t BUF_BYTES = (size_t)V_NODES * NCOL * 2;   // 51.2 MB per bf16 plane
  unsigned short* bufs[5];
  for (int i = 0; i < 5; ++i) bufs[i] = (unsigned short*)(ws + (size_t)i * BUF_BYTES);
  size_t off = 5 * BUF_BYTES;
  int*   row_ptr = (int*)(ws + off);   off += align256(sizeof(int) * (V_NODES + 1));
  int*   cursor  = (int*)(ws + off);   off += align256(sizeof(int) * V_NODES);
  int*   csr_src = (int*)(ws + off);   off += align256(sizeof(int) * E_EDGES);
  float* csr_w   = (float*)(ws + off); off += align256(sizeof(float) * E_EDGES);
  unsigned short* wbf = (unsigned short*)(ws + off); off += align256(sizeof(short) * 6 * 1024);
  float* stats   = (float*)(ws + off); off += 256;
  float* pooled  = (float*)(ws + off); off += 1024;

  // 1) transpose input to bf16 (V,256) plane, col = b*16+c; convert weights once
  dim3 tgrid(V_NODES / 16, NCOL / 16);
  k_transpose<<<tgrid, 256, 0, stream>>>(x, bufs[0]);
  k_wcvt<<<(6 * 1024 + 255) / 256, 256, 0, stream>>>(weights, wbf, 6 * 1024);

  // 2) CSR by destination
  hipMemsetAsync(cursor, 0, sizeof(int) * V_NODES, stream);
  k_count<<<(E_EDGES + 255) / 256, 256, 0, stream>>>(edge_dst, cursor);
  k_scan<<<1, 1024, 0, stream>>>(cursor, row_ptr);
  hipMemsetAsync(cursor, 0, sizeof(int) * V_NODES, stream);
  k_fill<<<(E_EDGES + 255) / 256, 256, 0, stream>>>(edge_src, edge_dst, edge_w, row_ptr,
                                                    cursor, csr_src, csr_w);

  unsigned short *P = bufs[0], *T1 = bufs[1], *T2 = bufs[2], *T3 = bufs[3], *H = bufs[4];

  auto lap = [&](unsigned short* y, const unsigned short* xx, const unsigned short* z,
                 float al, float be) {
    k_lap<<<V_NODES / 4, 256, 0, stream>>>(y, xx, z, al, be, row_ptr, csr_src, csr_w);
  };
  auto bn = [&](unsigned short* xx, int bni) {
    hipMemsetAsync(stats, 0, sizeof(float) * 64, stream);
    k_bn_stats<<<512, 256, 0, stream>>>(xx, stats);
    k_bn_fin<<<1, 16, 0, stream>>>(stats, gamma + (size_t)bni * 16, beta + (size_t)bni * 16);
    k_bn_apply<<<(V_NODES * NCOL / 4 + 255) / 256, 256, 0, stream>>>(xx, stats);
  };

  for (int i = 0; i < 3; ++i) {
    // h = relu(cheb(P, W[2i]))
    lap(T1, P, nullptr, 1.f, 0.f);          // x1 = L P
    lap(T2, T1, P, 2.f, -1.f);              // x2 = 2 L x1 - x0
    lap(T3, T2, T1, 2.f, -1.f);             // x3 = 2 L x2 - x1
    k_combine<false, true><<<TILES / 8, 256, 0, stream>>>(
        H, P, T1, T2, T3, wbf + (size_t)(2 * i) * 1024, biases + (size_t)(2 * i) * 16, nullptr);
    bn(H, i);                               // internal BN (gamma/beta row i)
    // out = relu(cheb(H, W[2i+1]) + P), written in place over T1 (row-local, safe)
    lap(T1, H, nullptr, 1.f, 0.f);
    lap(T2, T1, H, 2.f, -1.f);
    lap(T3, T2, T1, 2.f, -1.f);
    k_combine<true, true><<<TILES / 8, 256, 0, stream>>>(
        T1, H, T1, T2, T3, wbf + (size_t)(2 * i + 1) * 1024, biases + (size_t)(2 * i + 1) * 16, P);
    unsigned short* tmp = P; P = T1; T1 = tmp;   // rotate: output becomes next input
    if (i < 2) bn(P, 3 + i);                // inter-block BN (gamma/beta rows 3,4)
  }

  // max over V, softmax over channels
  hipMemsetAsync(pooled, 0, sizeof(float) * 256, stream);
  k_maxpool<<<256, 256, 0, stream>>>(P, pooled);
  k_softmax<<<1, 16, 0, stream>>>(pooled, out);
}
